// _PairEncodeCell_12661563588658
// MI455X (gfx1250) — matrix operand order repaired and checked by reading, not compiled, not run on hardware
//
#include <hip/hip_runtime.h>
#include <math.h>

// ---- problem constants (match reference) ----
#define TT 2048
#define BB 64
#define HH 256
#define AA 256
#define INN 512
#define NEGBIG (-1e12f)

// GEMM tiling
#define ROWS_PER_BLOCK 64      // 4 waves x 16 rows
#define LDSK 264               // 256 + 8 bf16 pad per row (avoids bank alignment)

typedef __attribute__((ext_vector_type(16))) __bf16 v16bf;
typedef __attribute__((ext_vector_type(8)))  float  v8f;

union Frag { uint4 q[2]; v16bf v; };

__device__ __forceinline__ unsigned short f2bf(float f) {
    unsigned int u = __float_as_uint(f);
    unsigned int r = u + 0x7FFFu + ((u >> 16) & 1u);   // round to nearest even
    return (unsigned short)(r >> 16);
}

__device__ __forceinline__ float sigmoidf_(float x) { return 1.0f / (1.0f + __expf(-x)); }

// branchless tanh: 1 - 2/(e^{2x}+1); saturates to +/-1 at +/-inf (2/inf -> 0)
__device__ __forceinline__ float fast_tanh(float x) {
    float t = __expf(2.0f * x);
    return 1.0f - __fdividef(2.0f, t + 1.0f);
}

// ---------------------------------------------------------------------------
// K0a: qs[b][a] = sum_k [input|state][b][k] * W_a1[k][a]   (k < 2H)
// ---------------------------------------------------------------------------
__global__ void k_qs(const float* __restrict__ input, const float* __restrict__ state,
                     const float* __restrict__ W_a1, float* __restrict__ qs) {
    int b = blockIdx.x;        // 64 blocks
    int a = threadIdx.x;       // 256 threads
    float acc = 0.f;
    for (int k = 0; k < HH; ++k)
        acc += input[b * HH + k] * W_a1[k * AA + a];
    for (int k = 0; k < HH; ++k)
        acc += state[b * HH + k] * W_a1[(HH + k) * AA + a];
    qs[b * AA + a] = acc;
}

// ---------------------------------------------------------------------------
// K0b: w2t[a*256 + h] = bf16( W_a1[(2H + h)*A + a] )  -- N-major, K-contiguous
// ---------------------------------------------------------------------------
__global__ void k_w2t(const float* __restrict__ W_a1, unsigned short* __restrict__ w2t) {
    int idx = blockIdx.x * blockDim.x + threadIdx.x;   // 65536 total
    int h = idx & 255;
    int a = idx >> 8;
    w2t[a * HH + h] = f2bf(W_a1[(2 * HH + h) * AA + a]);
}

// ---------------------------------------------------------------------------
// K1: logits[t*B+b] = mask ? sum_a tanh(qs[b][a] + (memory @ W2)[tb][a]) * w_a2[a]
//                          : -1e12
// bf16 WMMA GEMM: M = T*B rows, K = H = 256, N = A = 256.
// Block: 128 threads = 4 waves; block covers 64 rows; wave covers 16 rows.
// ---------------------------------------------------------------------------
__global__ void k_logits(const float* __restrict__ memory,
                         const float* __restrict__ qs,
                         const unsigned short* __restrict__ w2t,
                         const float* __restrict__ w_a2,
                         const unsigned char* __restrict__ mask,  // bool [T,B]
                         float* __restrict__ logits) {
    __shared__ unsigned short As[ROWS_PER_BLOCK * LDSK];

    const int tid  = threadIdx.x;
    const int lane = tid & 31;
    const int wave = tid >> 5;
    const int m    = lane & 15;
    const int half = lane >> 4;
    const int blockRow0 = blockIdx.x * ROWS_PER_BLOCK;

    // ---- stage: load 64x256 fp32 tile, convert to bf16 in LDS ----
    {
        const float4* gm = (const float4*)(memory + (size_t)blockRow0 * HH);
        #pragma unroll
        for (int i = 0; i < 32; ++i) {
            int flat = (i * 128 + tid) * 4;       // element index in 64x256 tile
            float4 d = gm[i * 128 + tid];
            int row = flat >> 8;
            int col = flat & 255;
            unsigned long long p =
                (unsigned long long)f2bf(d.x)        |
                ((unsigned long long)f2bf(d.y) << 16) |
                ((unsigned long long)f2bf(d.z) << 32) |
                ((unsigned long long)f2bf(d.w) << 48);
            *(unsigned long long*)&As[row * LDSK + col] = p;
        }
    }
    __syncthreads();

    // ---- hoist A fragments (8 K-steps), shared across all 16 column tiles ----
    Frag fa[8];
    #pragma unroll
    for (int ks = 0; ks < 8; ++ks) {
        const unsigned short* ap = &As[(wave * 16 + m) * LDSK + ks * 32 + half * 8];
        fa[ks].q[0] = *(const uint4*)(ap);        // K = kb + half*8 .. +7
        fa[ks].q[1] = *(const uint4*)(ap + 16);   // K = kb + 16 + half*8 .. +7
    }

    const int rowbase = blockRow0 + wave * 16;
    float psum[8];
    #pragma unroll
    for (int v = 0; v < 8; ++v) psum[v] = 0.f;

    for (int ct = 0; ct < 16; ++ct) {
        const int colbase = ct * 16;
        v8f acc = {0.f, 0.f, 0.f, 0.f, 0.f, 0.f, 0.f, 0.f};
        #pragma unroll
        for (int ks = 0; ks < 8; ++ks) {
            const unsigned short* bp = &w2t[(colbase + m) * HH + ks * 32 + half * 8];
            Frag fb;
            fb.q[0] = *(const uint4*)(bp);
            fb.q[1] = *(const uint4*)(bp + 16);
            acc = __builtin_amdgcn_wmma_f32_16x16x32_bf16(
                    false, fa[ks].v, false, fb.v, (short)0, acc, false, false);
        }
        // epilogue: tanh(c + qs) * w_a2, accumulate per-row partial (branchless)
        const int n   = colbase + m;
        const float w = w_a2[n];
        #pragma unroll
        for (int v = 0; v < 8; ++v) {
            int grow = rowbase + v + 8 * half;   // global row = t*B + b
            int b    = grow & (BB - 1);
            psum[v] += fast_tanh(acc[v] + qs[b * AA + n]) * w;
        }
    }

    // reduce the 16 column-lanes within each half-wave
    #pragma unroll
    for (int off = 1; off < 16; off <<= 1) {
        #pragma unroll
        for (int v = 0; v < 8; ++v)
            psum[v] += __shfl_xor(psum[v], off, 32);
    }
    if (m == 0) {
        #pragma unroll
        for (int v = 0; v < 8; ++v) {
            int grow = rowbase + v + 8 * half;
            logits[grow] = mask[grow] ? psum[v] : NEGBIG;
        }
    }
}

// ---------------------------------------------------------------------------
// K2: softmax over T per batch column (in place: logits -> scores)
// ---------------------------------------------------------------------------
__global__ void k_softmax(float* __restrict__ logits) {
    __shared__ float red[256];
    const int b = blockIdx.x;
    const int tid = threadIdx.x;

    float mx = -3.0e38f;
    for (int t = tid; t < TT; t += 256) mx = fmaxf(mx, logits[t * BB + b]);
    red[tid] = mx; __syncthreads();
    for (int o = 128; o > 0; o >>= 1) {
        if (tid < o) red[tid] = fmaxf(red[tid], red[tid + o]);
        __syncthreads();
    }
    float mval = red[0]; __syncthreads();

    float s = 0.f;
    for (int t = tid; t < TT; t += 256) s += __expf(logits[t * BB + b] - mval);
    red[tid] = s; __syncthreads();
    for (int o = 128; o > 0; o >>= 1) {
        if (tid < o) red[tid] += red[tid + o];
        __syncthreads();
    }
    float inv = 1.0f / red[0]; __syncthreads();

    for (int t = tid; t < TT; t += 256)
        logits[t * BB + b] = __expf(logits[t * BB + b] - mval) * inv;
}

// ---------------------------------------------------------------------------
// K3: chunked attn partials: attnp[c][b][h] = sum_{t in chunk c} s[t,b]*mem[t,b,h]
// (fixed-order partials + separate reduce -> deterministic, no float atomics)
// ---------------------------------------------------------------------------
#define TCHUNKS 16
__global__ void k_attn(const float* __restrict__ scores, const float* __restrict__ memory,
                       float* __restrict__ attnp) {
    const int b = blockIdx.x;
    const int c = blockIdx.y;
    const int h = threadIdx.x;
    const int tl = TT / TCHUNKS;   // 128
    float acc = 0.f;
    for (int i = 0; i < tl; ++i) {
        int t = c * tl + i;
        acc += scores[t * BB + b] * memory[((size_t)t * BB + b) * HH + h];
    }
    attnp[(c * BB + b) * HH + h] = acc;
}

__global__ void k_attnred(const float* __restrict__ attnp, float* __restrict__ attn) {
    const int b = blockIdx.x;
    const int h = threadIdx.x;
    float acc = 0.f;
    #pragma unroll
    for (int c = 0; c < TCHUNKS; ++c)
        acc += attnp[(c * BB + b) * HH + h];
    attn[b * HH + h] = acc;
}

// ---------------------------------------------------------------------------
// K4a: gated[b][j] = new_input[b][j] * sigmoid( (new_input @ W_gate)[b][j] )
// ---------------------------------------------------------------------------
__global__ void k_gate(const float* __restrict__ input, const float* __restrict__ attn,
                       const float* __restrict__ W_gate, float* __restrict__ gated) {
    int idx = blockIdx.x * blockDim.x + threadIdx.x;   // 64*512
    int b = idx >> 9;
    int j = idx & 511;
    float acc = 0.f;
    for (int k = 0; k < HH; ++k)  acc += input[b * HH + k] * W_gate[k * INN + j];
    for (int k = 0; k < HH; ++k)  acc += attn[b * HH + k]  * W_gate[(HH + k) * INN + j];
    float nij = (j < HH) ? input[b * HH + j] : attn[b * HH + (j - HH)];
    gated[b * INN + j] = nij * sigmoidf_(acc);
}

// ---------------------------------------------------------------------------
// K4b: GRU cell (torch gate order r,z,n)
// ---------------------------------------------------------------------------
__global__ void k_gru(const float* __restrict__ gated, const float* __restrict__ state,
                      const float* __restrict__ W_ih, const float* __restrict__ W_hh,
                      const float* __restrict__ b_ih, const float* __restrict__ b_hh,
                      float* __restrict__ out) {
    const int b = blockIdx.x;
    const int h = threadIdx.x;
    float ir = b_ih[h], iz = b_ih[HH + h], in_ = b_ih[2 * HH + h];
    for (int k = 0; k < INN; ++k) {
        float g = gated[b * INN + k];
        ir  += g * W_ih[(size_t)h * INN + k];
        iz  += g * W_ih[(size_t)(HH + h) * INN + k];
        in_ += g * W_ih[(size_t)(2 * HH + h) * INN + k];
    }
    float hr = b_hh[h], hz = b_hh[HH + h], hn = b_hh[2 * HH + h];
    for (int k = 0; k < HH; ++k) {
        float s = state[b * HH + k];
        hr += s * W_hh[h * HH + k];
        hz += s * W_hh[(HH + h) * HH + k];
        hn += s * W_hh[(2 * HH + h) * HH + k];
    }
    float r = sigmoidf_(ir + hr);
    float z = sigmoidf_(iz + hz);
    float n = tanhf(in_ + r * hn);
    out[b * HH + h] = (1.f - z) * n + z * state[b * HH + h];
}

// ---------------------------------------------------------------------------
extern "C" void kernel_launch(void* const* d_in, const int* in_sizes, int n_in,
                              void* d_out, int out_size, void* d_ws, size_t ws_size,
                              hipStream_t stream) {
    (void)in_sizes; (void)n_in; (void)out_size; (void)ws_size;

    const float*         input  = (const float*)d_in[0];
    const float*         memory = (const float*)d_in[1];
    const unsigned char* mask   = (const unsigned char*)d_in[2];  // bool [T,B]
    const float*         state  = (const float*)d_in[3];
    const float*         W_a1   = (const float*)d_in[4];
    const float*         w_a2   = (const float*)d_in[5];
    const float*         W_gate = (const float*)d_in[6];
    const float*         W_ih   = (const float*)d_in[7];
    const float*         W_hh   = (const float*)d_in[8];
    const float*         b_ih   = (const float*)d_in[9];
    const float*         b_hh   = (const float*)d_in[10];
    float* out = (float*)d_out;

    // workspace carve-up (~1.9 MB total)
    char* ws = (char*)d_ws;
    unsigned short* w2t    = (unsigned short*)ws;                 // 131072 B
    float*          qs     = (float*)(ws + 131072);               //  65536 B
    float*          logits = (float*)(ws + 131072 + 65536);       // 524288 B
    float*          attnp  = (float*)(ws + 131072 + 65536 + 524288);            // 1 MB
    float*          attn   = (float*)(ws + 131072 + 65536 + 524288 + 1048576);  // 64 KB
    float*          gated  = (float*)(ws + 131072 + 65536 + 524288 + 1048576 + 65536); // 128 KB

    k_qs    <<<BB, AA, 0, stream>>>(input, state, W_a1, qs);
    k_w2t   <<<(HH * AA) / 256, 256, 0, stream>>>(W_a1, w2t);
    k_logits<<<(TT * BB) / ROWS_PER_BLOCK, 128, 0, stream>>>(memory, qs, w2t, w_a2, mask, logits);
    k_softmax<<<BB, 256, 0, stream>>>(logits);
    k_attn  <<<dim3(BB, TCHUNKS), HH, 0, stream>>>(logits, memory, attnp);
    k_attnred<<<BB, HH, 0, stream>>>(attnp, attn);
    k_gate  <<<(BB * INN) / 256, 256, 0, stream>>>(input, attn, W_gate, gated);
    k_gru   <<<BB, HH, 0, stream>>>(gated, state, W_ih, W_hh, b_ih, b_hh, out);
}
